// ScouterAttention_28638841930203
// MI455X (gfx1250) — compile-verified, matrix operand order checked by reading, not run on hardware
//
#include <hip/hip_runtime.h>
#include <hip/hip_bf16.h>

// ---------------------------------------------------------------------------
// ScouterAttention for MI455X (gfx1250, wave32, WMMA).
// B=64, N=4096, D=128, S=50. All GEMMs via v_wmma_f32_16x16x32_f16.
// Memory-bound problem (~0.7GB HBM traffic vs ~33 GFLOP), so: f16 data path
// with f32 accumulate, fully fused 3-layer MLP (weights + ping-pong tiles in
// 160KB dynamic LDS -- CDNA5 WGP has 320KB), single write of k as f16.
// ---------------------------------------------------------------------------

typedef _Float16 v16h __attribute__((ext_vector_type(16)));
typedef _Float16 v8h  __attribute__((ext_vector_type(8)));
typedef float    v8f  __attribute__((ext_vector_type(8)));

#define BB 64
#define NN 4096
#define DD 128
#define SS 50
#define SPAD 64          // S padded to 4 WMMA row-tiles
#define KSLICES 4        // deterministic K-split for the updates GEMM

// ---- WMMA wrapper: D = A(16x32 f16) * B(32x16 f16) + C(16x16 f32) ----------
__device__ __forceinline__ v8f wmma_f16(v16h a, v16h b, v8f c) {
  return __builtin_amdgcn_wmma_f32_16x16x32_f16(
      /*neg_a=*/false, a, /*neg_b=*/false, b,
      /*c_mod=*/(short)0, c, /*reuse_a=*/false, /*reuse_b=*/false);
}

// ---- A-fragment (16x32, f16) from a row-major [16 x ld] strip --------------
// ISA 7.12.2: lane r=lane&15 holds row r; lanes 0-15 carry K 0..7 & 16..23,
// lanes 16-31 carry K 8..15 & 24..31. Two contiguous 16B loads per lane.
__device__ __forceinline__ v16h frag_ld_f16(const _Float16* base, int ld) {
  const int lane = threadIdx.x & 31;
  const int r    = lane & 15;
  const int k0   = (lane & 16) ? 8 : 0;
  const _Float16* p = base + (size_t)r * ld + k0;
  union { v16h v; v8h h2[2]; } u;
  u.h2[0] = *(const v8h*)(p);
  u.h2[1] = *(const v8h*)(p + 16);
  return u.v;
}

// Same, but rows >= rowsValid return zeros (S=50 padded to 64).
__device__ __forceinline__ v16h frag_ld_f16_guard(const _Float16* base, int ld,
                                                  int rowsValid) {
  const int lane = threadIdx.x & 31;
  const int r    = lane & 15;
  const int k0   = (lane & 16) ? 8 : 0;
  union { v16h v; v8h h2[2]; } u;
  if (r < rowsValid) {
    const _Float16* p = base + (size_t)r * ld + k0;
    u.h2[0] = *(const v8h*)(p);
    u.h2[1] = *(const v8h*)(p + 16);
  } else {
    u.h2[0] = (v8h)(_Float16)0.f;
    u.h2[1] = (v8h)(_Float16)0.f;
  }
  return u.v;
}

// A-fragment from f32 row-major source, scaled + converted on the fly,
// row-guarded (used for q = slots_init with the 1/sqrt(D) folded in).
__device__ __forceinline__ v16h frag_ld_a_f32(const float* base, int ld,
                                              int rowsValid, float scl) {
  const int lane = threadIdx.x & 31;
  const int r    = lane & 15;
  const int k0   = (lane & 16) ? 8 : 0;
  union { v16h v; _Float16 h[16]; } u;
  if (r < rowsValid) {
    const float* p = base + (size_t)r * ld + k0;
#pragma unroll
    for (int j = 0; j < 8; ++j) u.h[j]     = (_Float16)(p[j]      * scl);
#pragma unroll
    for (int j = 0; j < 8; ++j) u.h[8 + j] = (_Float16)(p[16 + j] * scl);
  } else {
#pragma unroll
    for (int j = 0; j < 16; ++j) u.h[j] = (_Float16)0.f;
  }
  return u.v;
}

// B-fragment (32x16) from a row-major f32 [K x ld] tile (lane = column),
// converted to f16 on the fly (used for inputs_x in the updates GEMM).
__device__ __forceinline__ v16h frag_ld_b_f32(const float* base, int ld) {
  const int lane = threadIdx.x & 31;
  const int n    = lane & 15;
  const int k0   = (lane & 16) ? 8 : 0;
  union { v16h v; _Float16 h[16]; } u;
#pragma unroll
  for (int j = 0; j < 8; ++j) u.h[j]     = (_Float16)base[(size_t)(k0 + j)      * ld + n];
#pragma unroll
  for (int j = 0; j < 8; ++j) u.h[8 + j] = (_Float16)base[(size_t)(k0 + 16 + j) * ld + n];
  return u.v;
}

// ---------------------------------------------------------------------------
// Kernel 1: fused 3-layer MLP. One block = 128 rows of [B*N, D].
// LDS: tileA(32K) + tileB(32K) + 3 transposed f16 weights (96K) = 160KB.
// ---------------------------------------------------------------------------
template <bool RELU>
__device__ __forceinline__ void mlp_layer(const _Float16* src, const _Float16* wT,
                                          const float* bias, _Float16* dst,
                                          size_t dstLd, size_t dstRow0, int m0) {
  const int lane = threadIdx.x & 31;
  v16h af[4];
#pragma unroll
  for (int kk = 0; kk < 4; ++kk)
    af[kk] = frag_ld_f16(src + (size_t)m0 * DD + kk * 32, DD);
#pragma unroll
  for (int nt = 0; nt < 8; ++nt) {
    v8f acc = {};
#pragma unroll
    for (int kk = 0; kk < 4; ++kk)
      acc = wmma_f16(af[kk], frag_ld_f16(wT + (size_t)(nt * 16) * DD + kk * 32, DD), acc);
    const int   n  = nt * 16 + (lane & 15);
    const float bv = bias[n];
    const int   mb = m0 + ((lane & 16) ? 8 : 0);
#pragma unroll
    for (int v = 0; v < 8; ++v) {
      float x = acc[v] + bv;
      if (RELU) x = fmaxf(x, 0.f);
      dst[(dstRow0 + (size_t)(mb + v)) * dstLd + n] = (_Float16)x;
    }
  }
}

__global__ void mlp_kernel(const float* __restrict__ in,
                           const float* __restrict__ W1, const float* __restrict__ b1,
                           const float* __restrict__ W2, const float* __restrict__ b2,
                           const float* __restrict__ W3, const float* __restrict__ b3,
                           _Float16* __restrict__ kout) {
  extern __shared__ char smemraw[];
  _Float16* tileA = (_Float16*)smemraw;        // 128x128 activations (ping)
  _Float16* tileB = tileA + DD * DD;           // 128x128 activations (pong)
  _Float16* wt    = tileB + DD * DD;           // 3 x 128x128, transposed [n][k]

  const int    tid  = threadIdx.x;
  const size_t row0 = (size_t)blockIdx.x * DD;

  __builtin_prefetch(in + row0 * DD + tid * 64, 0, 0);   // global_prefetch_b8

  // Input tile f32 -> f16 (vectorized float4 reads).
  for (int i = tid * 4; i < DD * DD; i += 256 * 4) {
    float4 f = *(const float4*)(in + row0 * DD + i);
    tileA[i + 0] = (_Float16)f.x;  tileA[i + 1] = (_Float16)f.y;
    tileA[i + 2] = (_Float16)f.z;  tileA[i + 3] = (_Float16)f.w;
  }
  // Weights: transpose into LDS so B-fragments load contiguously (ds_load_b128).
  {
    const float* Ws0 = W1; const float* Ws1 = W2; const float* Ws2 = W3;
    for (int i = tid; i < DD * DD; i += 256) {
      const int k = i >> 7, n = i & 127;
      wt[0 * DD * DD + n * DD + k] = (_Float16)Ws0[i];
      wt[1 * DD * DD + n * DD + k] = (_Float16)Ws1[i];
      wt[2 * DD * DD + n * DD + k] = (_Float16)Ws2[i];
    }
  }
  __syncthreads();

  const int m0 = (tid >> 5) * 16;   // 8 waves -> 8 row strips of 16

  mlp_layer<true >(tileA, wt + 0 * DD * DD, b1, tileB, DD, 0, m0);
  __syncthreads();
  mlp_layer<true >(tileB, wt + 1 * DD * DD, b2, tileA, DD, 0, m0);
  __syncthreads();
  mlp_layer<false>(tileA, wt + 2 * DD * DD, b3, kout, DD, row0, m0);
}

// ---------------------------------------------------------------------------
// Kernel 2: dots[b,s,n] = scale * q . k[b,n,:]   (scale folded into q frags).
// k is [tok, d] row-major = B^T, so B-fragments use the contiguous A-loader.
// Block: 8 waves, batch b = blockIdx.y, 128 tokens per block.
// ---------------------------------------------------------------------------
__global__ void dots_kernel(const float* __restrict__ slots,
                            const _Float16* __restrict__ k16,
                            float* __restrict__ dots /* [B][S][N] */) {
  const int b      = blockIdx.y;
  const int tok0   = blockIdx.x * 128;
  const int wave   = threadIdx.x >> 5;
  const int lane   = threadIdx.x & 31;
  const int mi     = wave & 3;
  const int ntBase = (wave >> 2) * 4;
  const int m0     = mi * 16;
  const float scl  = 0.088388347648318447f;   // 1/sqrt(128)

  v16h af[4];
#pragma unroll
  for (int kk = 0; kk < 4; ++kk)
    af[kk] = frag_ld_a_f32(slots + (size_t)m0 * DD + kk * 32, DD, SS - m0, scl);

  const _Float16* kbase = k16 + ((size_t)b * NN + tok0) * DD;
#pragma unroll
  for (int t = 0; t < 4; ++t) {
    const int nt = ntBase + t;
    v8f acc = {};
#pragma unroll
    for (int kk = 0; kk < 4; ++kk)
      acc = wmma_f16(af[kk], frag_ld_f16(kbase + (size_t)(nt * 16) * DD + kk * 32, DD), acc);
    const int n  = tok0 + nt * 16 + (lane & 15);
    const int mb = m0 + ((lane & 16) ? 8 : 0);
#pragma unroll
    for (int v = 0; v < 8; ++v) {
      const int m = mb + v;
      if (m < SS) dots[((size_t)b * SS + m) * NN + n] = acc[v];
    }
  }
}

// ---------------------------------------------------------------------------
// Kernel 3/4: row sums over tokens, then totals over slots.
// ---------------------------------------------------------------------------
__global__ void rowsum_kernel(const float* __restrict__ dots, float* __restrict__ rowsum) {
  __shared__ float red[256];
  const float* row = dots + (size_t)blockIdx.x * NN;
  float s = 0.f;
  for (int i = threadIdx.x; i < NN; i += 256) s += row[i];
  red[threadIdx.x] = s; __syncthreads();
  for (int off = 128; off > 0; off >>= 1) {
    if ((int)threadIdx.x < off) red[threadIdx.x] += red[threadIdx.x + off];
    __syncthreads();
  }
  if (threadIdx.x == 0) rowsum[blockIdx.x] = red[0];
}

__global__ void total_kernel(const float* __restrict__ rowsum, float* __restrict__ total) {
  __shared__ float red[64];
  const int b = blockIdx.x;
  red[threadIdx.x] = ((int)threadIdx.x < SS) ? rowsum[b * SS + threadIdx.x] : 0.f;
  __syncthreads();
  for (int off = 32; off > 0; off >>= 1) {
    if ((int)threadIdx.x < off) red[threadIdx.x] += red[threadIdx.x + off];
    __syncthreads();
  }
  if (threadIdx.x == 0) total[b] = red[0];
}

// ---------------------------------------------------------------------------
// Kernel 5: renorm + softmax(axis=slots) * sigmoid. One thread per (b,n)
// column; loads are coalesced across n. Writes attn f32 in place (d_out) and
// an f16 copy used as the A matrix of the updates GEMM.
// ---------------------------------------------------------------------------
__global__ void softmax_kernel(float* __restrict__ dots,
                               const float* __restrict__ rowsum,
                               const float* __restrict__ total,
                               _Float16* __restrict__ attn16) {
  const int g = blockIdx.x * 256 + threadIdx.x;   // over B*N
  const int b = g >> 12;                          // / 4096
  const int n = g & (NN - 1);
  const float tot = total[b];

  float v[SS];
  float mx = -3.4e38f;
#pragma unroll
  for (int s = 0; s < SS; ++s) {
    float d = dots[((size_t)b * SS + s) * NN + n];
    d = d / rowsum[b * SS + s] * tot;
    v[s] = d;
    mx = fmaxf(mx, d);
  }
  float sum = 0.f;
#pragma unroll
  for (int s = 0; s < SS; ++s) sum += __expf(v[s] - mx);
  const float inv = 1.f / sum;
#pragma unroll
  for (int s = 0; s < SS; ++s) {
    const float sm  = __expf(v[s] - mx) * inv;
    const float sig = 1.f / (1.f + __expf(-v[s]));
    const float a   = sm * sig;
    dots[((size_t)b * SS + s) * NN + n]   = a;
    attn16[((size_t)b * SS + s) * NN + n] = (_Float16)a;
  }
}

// ---------------------------------------------------------------------------
// Kernel 6: updates partials. updates[b,s,d] = sum_tok attn[b,s,tok]*x[b,tok,d].
// K=4096 split over KSLICES blocks (deterministic partial buffers, no atomics).
// A = attn16 (row-guarded at S=50), B = x f32 converted on the fly.
// ---------------------------------------------------------------------------
__global__ void updates_partial(const _Float16* __restrict__ attn16,
                                const float* __restrict__ x,
                                float* __restrict__ part /* [KS][B][64][128] */) {
  const int slice  = blockIdx.x;
  const int b      = blockIdx.y;
  const int wave   = threadIdx.x >> 5;
  const int lane   = threadIdx.x & 31;
  const int mi     = wave & 3;
  const int ntBase = (wave >> 2) * 4;
  const int m0     = mi * 16;
  const int tokBase = slice * (NN / KSLICES);

  __builtin_prefetch(x + ((size_t)b * NN + tokBase) * DD + threadIdx.x * 64, 0, 0);

  v8f acc[4] = {};
  for (int ks = 0; ks < (NN / KSLICES) / 32; ++ks) {
    const int tok0 = tokBase + ks * 32;
    const v16h a = frag_ld_f16_guard(attn16 + ((size_t)b * SS + m0) * NN + tok0, NN, SS - m0);
#pragma unroll
    for (int t = 0; t < 4; ++t) {
      const int nt = ntBase + t;
      const v16h bf = frag_ld_b_f32(x + ((size_t)b * NN + tok0) * DD + nt * 16, DD);
      acc[t] = wmma_f16(a, bf, acc[t]);
    }
  }
  const int mb = m0 + ((lane & 16) ? 8 : 0);
#pragma unroll
  for (int t = 0; t < 4; ++t) {
    const int n = (ntBase + t) * 16 + (lane & 15);
#pragma unroll
    for (int v = 0; v < 8; ++v)
      part[((size_t)(slice * BB + b) * SPAD + (mb + v)) * DD + n] = acc[t][v];
  }
}

__global__ void updates_reduce(const float* __restrict__ part, float* __restrict__ out) {
  const int g  = blockIdx.x * 256 + threadIdx.x;   // over B*S*D = 409600
  const int d  = g & (DD - 1);
  const int bs = g >> 7;
  const int b  = bs / SS;
  const int s  = bs - b * SS;
  float acc = 0.f;
#pragma unroll
  for (int sl = 0; sl < KSLICES; ++sl)
    acc += part[((size_t)(sl * BB + b) * SPAD + s) * DD + d];
  out[g] = acc * (1.0f / (float)DD);
}

// ---------------------------------------------------------------------------
// Host launcher
// ---------------------------------------------------------------------------
extern "C" void kernel_launch(void* const* d_in, const int* in_sizes, int n_in,
                              void* d_out, int out_size, void* d_ws, size_t ws_size,
                              hipStream_t stream) {
  const float* inputs   = (const float*)d_in[0];
  const float* inputs_x = (const float*)d_in[1];
  const float* slots    = (const float*)d_in[2];
  const float* W1 = (const float*)d_in[3];  const float* b1 = (const float*)d_in[4];
  const float* W2 = (const float*)d_in[5];  const float* b2 = (const float*)d_in[6];
  const float* W3 = (const float*)d_in[7];  const float* b3 = (const float*)d_in[8];

  float* out_updates = (float*)d_out;                               // [B,S,D]
  float* dots        = out_updates + (size_t)BB * SS * DD;          // [B,S,N] (attn in place)

  // Workspace layout (bytes)
  char* ws = (char*)d_ws;
  const size_t k16_bytes    = (size_t)BB * NN * DD * 2;   // 67,108,864
  const size_t attn16_bytes = (size_t)BB * SS * NN * 2;   // 26,214,400
  _Float16* k16    = (_Float16*)ws;
  _Float16* attn16 = (_Float16*)(ws + k16_bytes);
  float*    rowsum = (float*)(ws + k16_bytes + attn16_bytes);
  float*    total  = rowsum + (size_t)BB * SS;
  float*    part   = (float*)(((uintptr_t)(total + BB) + 255) & ~(uintptr_t)255);

  // 160KB dynamic LDS for the fused MLP (CDNA5: 320KB LDS per WGP)
  const int mlp_smem = (2 * DD * DD + 3 * DD * DD) * (int)sizeof(_Float16);  // 163840
  hipFuncSetAttribute((const void*)mlp_kernel,
                      hipFuncAttributeMaxDynamicSharedMemorySize, mlp_smem);

  mlp_kernel<<<dim3((BB * NN) / 128), 256, mlp_smem, stream>>>(
      inputs, W1, b1, W2, b2, W3, b3, k16);

  dots_kernel<<<dim3(NN / 128, BB), 256, 0, stream>>>(slots, k16, dots);

  rowsum_kernel<<<dim3(BB * SS), 256, 0, stream>>>(dots, rowsum);
  total_kernel<<<dim3(BB), 64, 0, stream>>>(rowsum, total);

  softmax_kernel<<<dim3((BB * NN) / 256), 256, 0, stream>>>(dots, rowsum, total, attn16);

  updates_partial<<<dim3(KSLICES, BB), 256, 0, stream>>>(attn16, inputs_x, part);
  updates_reduce<<<dim3((BB * SS * DD) / 256), 256, 0, stream>>>(part, out_updates);
}
